// TriangleMultiplication_927712936257
// MI455X (gfx1250) — matrix operand order repaired and checked by reading, not compiled, not run on hardware
//
#include <hip/hip_runtime.h>
#include <stdint.h>

// Problem constants
constexpr int kB  = 2;
constexpr int kN  = 384;
constexpr int kD  = 128;   // input dim
constexpr int kDH = 128;   // hidden dim
constexpr float kEps = 1e-5f;

typedef float          v8f    __attribute__((ext_vector_type(8)));
typedef __bf16         v16bf  __attribute__((ext_vector_type(16)));
typedef unsigned short u16x8  __attribute__((ext_vector_type(8)));
typedef unsigned short u16x16 __attribute__((ext_vector_type(16)));

union BF16Frag { u16x16 u; v16bf b; };

__device__ __forceinline__ unsigned short f2bf(float f) {
  union { float f; unsigned u; } v; v.f = f;
  unsigned r = v.u + 0x7FFFu + ((v.u >> 16) & 1u);   // round-to-nearest-even
  return (unsigned short)(r >> 16);
}
__device__ __forceinline__ float bf2f(unsigned short h) {
  union { unsigned u; float f; } v; v.u = ((unsigned)h) << 16;
  return v.f;
}
__device__ __forceinline__ float sigmoidf_fast(float x) {
  return 1.0f / (1.0f + __expf(-x));
}
__device__ __forceinline__ v16bf make_frag(u16x8 lo, u16x8 hi) {
  BF16Frag f;
#pragma unroll
  for (int i = 0; i < 8; ++i) { f.u[i] = lo[i]; f.u[8 + i] = hi[i]; }
  return f.b;
}

// ---------------------------------------------------------------------------
// Kernel 0: convert weights to bf16, transposed so WMMA B-fragments are
// two contiguous 8-half K-runs per lane.
//   wt   [640][128] : rows 0..511 = W_lr^T, rows 512..639 = W_gate^T
//   woutt[128][128] : W_out^T
// ---------------------------------------------------------------------------
__global__ __launch_bounds__(256)
void prep_weights_kernel(const float* __restrict__ Wlr,
                         const float* __restrict__ Wgate,
                         const float* __restrict__ Wout,
                         unsigned short* __restrict__ wt,
                         unsigned short* __restrict__ woutt) {
  int idx = blockIdx.x * 256 + threadIdx.x;
  const int total_wt = 640 * 128;
  if (idx < total_wt) {
    int o = idx >> 7, i = idx & 127;
    float v = (o < 512) ? Wlr[i * 512 + o] : Wgate[i * 128 + (o - 512)];
    wt[o * 128 + i] = f2bf(v);
  } else {
    int j = idx - total_wt;   // < 128*128 by grid sizing
    int o = j >> 7, i = j & 127;
    woutt[o * 128 + i] = f2bf(Wout[i * 128 + o]);
  }
}

// ---------------------------------------------------------------------------
// Kernel A: fused projection. Block = (b, q, 32-wide k-tile).
//   h[32][640] = bf16(x[b, k0..k0+32, q, :]) @ [W_lr | W_gate]   (WMMA)
//   GLU + mask + sigmoid epilogue; store left/right transposed to
//   [b][d][q][k] (bf16) and gate to [b][k][q][d] (bf16), vectorized u16x8.
// ---------------------------------------------------------------------------
__global__ __launch_bounds__(256)
void proj_glu_kernel(const float* __restrict__ x,
                     const unsigned char* __restrict__ mask,
                     const unsigned short* __restrict__ wt,    // [640][128] bf16
                     unsigned short* __restrict__ leftT,       // [B][DH][N][N]
                     unsigned short* __restrict__ rightT,      // [B][DH][N][N]
                     unsigned short* __restrict__ gateB) {     // [B][N][N][DH]
  __shared__ unsigned short As[32][136];   // x tile bf16 [k][d], padded
  __shared__ float          hs[32][644];   // h tile fp32 [k][c], padded
  __shared__ float          mrow[32];      // mask[b, k0+r] as float

  const int b    = blockIdx.z;
  const int q    = blockIdx.y;
  const int k0   = blockIdx.x * 32;
  const int t    = threadIdx.x;
  const int lane = t & 31;
  const int wv   = t >> 5;
  const int l15  = lane & 15;
  const int hi   = (lane < 16) ? 0 : 1;

  // Stage x tile as bf16 (coalesced 128-float runs) + mask row.
  if (t < 32) mrow[t] = (float)mask[b * kN + k0 + t];
#pragma unroll
  for (int it = 0; it < 16; ++it) {
    int idx = t + 256 * it;
    int r = idx >> 7, d = idx & 127;
    float v = x[(((size_t)b * kN + (k0 + r)) * kN + q) * kD + d];
    As[r][d] = f2bf(v);
  }
  __syncthreads();

  // 2 x 40 output tiles of 16x16, 10 per wave, K = 128 (4 WMMA each).
  for (int T = wv; T < 80; T += 8) {
    int mt = T & 1, nt = T >> 1;
    int mrow_a = 16 * mt + l15;
    int ncol   = 16 * nt + l15;
    v8f acc = {};
#pragma unroll
    for (int kk = 0; kk < 4; ++kk) {
      int k = kk * 32;
      u16x8 a0 = *(const u16x8*)&As[mrow_a][k + hi * 8];
      u16x8 a1 = *(const u16x8*)&As[mrow_a][k + hi * 8 + 16];
      v16bf Af = make_frag(a0, a1);
      BF16Frag Bf;
      Bf.b = make_frag(*(const u16x8*)&wt[ncol * 128 + k + hi * 8],
                       *(const u16x8*)&wt[ncol * 128 + k + hi * 8 + 16]);
      acc = __builtin_amdgcn_wmma_f32_16x16x32_bf16(
          false, Af, false, Bf.b, (short)0, acc, false, false);
    }
#pragma unroll
    for (int r = 0; r < 8; ++r)
      hs[16 * mt + r + hi * 8][16 * nt + l15] = acc[r];
  }
  __syncthreads();

  // GLU + mask epilogue -> transposed bf16 stores, vectorized 8-wide over k.
  const float mq = (float)mask[b * kN + q];
#pragma unroll
  for (int it = 0; it < 2; ++it) {
    int chunk = t + 256 * it;                 // 512 chunks of 8 k-values
    int kc = chunk & 3, d = chunk >> 2;       // d: 0..127
    u16x8 lv8, rv8;
#pragma unroll
    for (int e = 0; e < 8; ++e) {
      int r = kc * 8 + e;
      float pm = mq * mrow[r];
      float lv = hs[r][d]       * sigmoidf_fast(hs[r][256 + d]) * pm;
      float rv = hs[r][128 + d] * sigmoidf_fast(hs[r][384 + d]) * pm;
      lv8[e] = f2bf(lv);
      rv8[e] = f2bf(rv);
    }
    size_t off = (((size_t)b * kDH + d) * kN + q) * kN + k0 + kc * 8;
    *(u16x8*)&leftT[off]  = lv8;
    *(u16x8*)&rightT[off] = rv8;
  }
#pragma unroll
  for (int it = 0; it < 2; ++it) {
    int chunk = t + 256 * it;                 // 512 chunks of 8 d-values
    int d0 = (chunk & 15) * 8, r = chunk >> 4;
    u16x8 g8;
#pragma unroll
    for (int e = 0; e < 8; ++e)
      g8[e] = f2bf(sigmoidf_fast(hs[r][512 + d0 + e]));
    *(u16x8*)&gateB[(((size_t)b * kN + (k0 + r)) * kN + q) * kDH + d0] = g8;
  }
}

// ---------------------------------------------------------------------------
// Kernel B: fused triangle matmul + LayerNorm + gate + output projection.
// One block per 16x16 (i,j) tile; wave w owns channels [16w, 16w+16).
//   tri[i,j,d] = sum_k rightT[b,d,i,k] * leftT[b,d,j,k]     (12 WMMA / chan)
// Next channel's rows are prefetched (global_prefetch_b8) during the WMMA
// chain. Mean/var accumulated in registers, cross-wave reduced in LDS,
// bf16 tile normalized in place (vectorized), then WMMA out-projection,
// multiplied by the gate, stored fp32.
// ---------------------------------------------------------------------------
__global__ __launch_bounds__(256)
void tri_ln_out_kernel(const unsigned short* __restrict__ leftT,
                       const unsigned short* __restrict__ rightT,
                       const unsigned short* __restrict__ gateB,
                       const unsigned short* __restrict__ woutt,  // [128][128]
                       const float* __restrict__ gamma,
                       const float* __restrict__ beta,
                       float* __restrict__ out) {
  __shared__ unsigned short nraw[256][136];   // bf16 [pos][d], padded
  __shared__ float s1[8][256];
  __shared__ float s2[8][256];

  const int b    = blockIdx.z;
  const int i0   = blockIdx.y * 16;
  const int j0   = blockIdx.x * 16;
  const int t    = threadIdx.x;
  const int lane = t & 31;
  const int wv   = t >> 5;
  const int l15  = lane & 15;
  const int hi   = (lane < 16) ? 0 : 1;

  float rs1[8], rs2[8];
#pragma unroll
  for (int r = 0; r < 8; ++r) { rs1[r] = 0.f; rs2[r] = 0.f; }

  const size_t bd0 = (size_t)b * kDH;
  const size_t rowR = (size_t)(i0 + l15);
  const size_t rowL = (size_t)(j0 + l15);

  // Warm first channel's rows.
  {
    int d = 16 * wv;
    const unsigned short* R0 = rightT + ((bd0 + d) * kN + rowR) * kN;
    const unsigned short* L0 = leftT  + ((bd0 + d) * kN + rowL) * kN;
#pragma unroll
    for (int pLine = 0; pLine < 3; ++pLine) {
      __builtin_prefetch(R0 + pLine * 128, 0, 3);
      __builtin_prefetch(L0 + pLine * 128, 0, 3);
    }
  }

  for (int c = 0; c < 16; ++c) {
    int d = 16 * wv + c;
    const unsigned short* Rbase = rightT + ((bd0 + d) * kN + rowR) * kN;
    const unsigned short* Lbase = leftT  + ((bd0 + d) * kN + rowL) * kN;

    // Prefetch next channel's rows (3 x 256B lines cover a 768B row).
    if (c < 15) {
      const unsigned short* Rn = Rbase + (size_t)kN * kN;
      const unsigned short* Ln = Lbase + (size_t)kN * kN;
#pragma unroll
      for (int pLine = 0; pLine < 3; ++pLine) {
        __builtin_prefetch(Rn + pLine * 128, 0, 3);
        __builtin_prefetch(Ln + pLine * 128, 0, 3);
      }
    }

    v8f acc = {};
#pragma unroll 4
    for (int kk = 0; kk < 12; ++kk) {
      int k = kk * 32;
      u16x8 a0 = *(const u16x8*)(Rbase + k + hi * 8);
      u16x8 a1 = *(const u16x8*)(Rbase + k + hi * 8 + 16);
      v16bf Af = make_frag(a0, a1);
      BF16Frag Bf;
      Bf.b = make_frag(*(const u16x8*)(Lbase + k + hi * 8),
                       *(const u16x8*)(Lbase + k + hi * 8 + 16));
      acc = __builtin_amdgcn_wmma_f32_16x16x32_bf16(
          false, Af, false, Bf.b, (short)0, acc, false, false);
    }
#pragma unroll
    for (int r = 0; r < 8; ++r) {
      int p = (r + hi * 8) * 16 + l15;
      nraw[p][d] = f2bf(acc[r]);
      rs1[r] += acc[r];
      rs2[r] += acc[r] * acc[r];
    }
  }
#pragma unroll
  for (int r = 0; r < 8; ++r) {
    int p = (r + hi * 8) * 16 + l15;
    s1[wv][p] = rs1[r];
    s2[wv][p] = rs2[r];
  }
  __syncthreads();

  // LayerNorm: one thread per (i,j) position; vectorized 8-wide over d.
  {
    float a = 0.f, bsq = 0.f;
#pragma unroll
    for (int w2 = 0; w2 < 8; ++w2) { a += s1[w2][t]; bsq += s2[w2][t]; }
    float mu   = a * (1.0f / kDH);
    float var  = bsq * (1.0f / kDH) - mu * mu;
    float rsig = rsqrtf(var + kEps);
#pragma unroll 4
    for (int d0 = 0; d0 < kDH; d0 += 8) {
      u16x8 v8 = *(const u16x8*)&nraw[t][d0];
      u16x8 o8;
#pragma unroll
      for (int e = 0; e < 8; ++e) {
        float v  = bf2f(v8[e]);
        float nv = (v - mu) * rsig * gamma[d0 + e] + beta[d0 + e];
        o8[e] = f2bf(nv);
      }
      *(u16x8*)&nraw[t][d0] = o8;
    }
  }
  __syncthreads();

  // Output projection: (256 pos x 128 dh) @ W_out^T, 16x8 tiles, 16/wave.
  for (int ti = 0; ti < 16; ++ti) {
    int T  = wv * 16 + ti;
    int mt = T >> 3, nt = T & 7;
    int prow = 16 * mt + l15;
    int ocol = 16 * nt + l15;
    v8f acc = {};
#pragma unroll
    for (int kk = 0; kk < 4; ++kk) {
      int k = kk * 32;
      u16x8 a0 = *(const u16x8*)&nraw[prow][k + hi * 8];
      u16x8 a1 = *(const u16x8*)&nraw[prow][k + hi * 8 + 16];
      v16bf Af = make_frag(a0, a1);
      BF16Frag Bf;
      Bf.b = make_frag(*(const u16x8*)&woutt[ocol * kDH + k + hi * 8],
                       *(const u16x8*)&woutt[ocol * kDH + k + hi * 8 + 16]);
      acc = __builtin_amdgcn_wmma_f32_16x16x32_bf16(
          false, Af, false, Bf.b, (short)0, acc, false, false);
    }
#pragma unroll
    for (int r = 0; r < 8; ++r) {
      int p = 16 * mt + r + hi * 8;
      int i = i0 + (p >> 4);
      int j = j0 + (p & 15);
      size_t off = (((size_t)b * kN + i) * kN + j) * kD + ocol;
      out[off] = acc[r] * bf2f(gateB[off]);
    }
  }
}

// ---------------------------------------------------------------------------
extern "C" void kernel_launch(void* const* d_in, const int* in_sizes, int n_in,
                              void* d_out, int out_size, void* d_ws, size_t ws_size,
                              hipStream_t stream) {
  (void)in_sizes; (void)n_in; (void)out_size; (void)ws_size;
  const float*         x     = (const float*)d_in[0];
  const unsigned char* mask  = (const unsigned char*)d_in[1];
  const float*         Wlr   = (const float*)d_in[2];
  const float*         Wgate = (const float*)d_in[3];
  const float*         gamma = (const float*)d_in[4];
  const float*         beta  = (const float*)d_in[5];
  const float*         Wout  = (const float*)d_in[6];
  float*               out   = (float*)d_out;

  // Workspace layout (all offsets 256B-aligned), bf16 buffers:
  //   wt    : 640*128            = 163,840 B
  //   woutt : 128*128            =  32,768 B
  //   leftT : B*DH*N*N           = 75,497,472 B
  //   rightT: B*DH*N*N           = 75,497,472 B
  //   gateB : B*N*N*DH           = 75,497,472 B   (total ~226.7 MB)
  char* ws = (char*)d_ws;
  unsigned short* wt     = (unsigned short*)(ws);
  unsigned short* woutt  = (unsigned short*)(ws + 163840);
  unsigned short* leftT  = (unsigned short*)(ws + 196608);
  unsigned short* rightT = (unsigned short*)(ws + 196608 + 75497472ull);
  unsigned short* gateB  = (unsigned short*)(ws + 196608 + 2ull * 75497472ull);

  prep_weights_kernel<<<384, 256, 0, stream>>>(Wlr, Wgate, Wout, wt, woutt);

  dim3 gA(kN / 32, kN, kB);          // (12, 384, 2)
  proj_glu_kernel<<<gA, 256, 0, stream>>>(x, mask, wt, leftT, rightT, gateB);

  dim3 gB(kN / 16, kN / 16, kB);     // (24, 24, 2)
  tri_ln_out_kernel<<<gB, 256, 0, stream>>>(leftT, rightT, gateB, woutt,
                                            gamma, beta, out);
}
